// MultiHeadAttention_19533511262319
// MI455X (gfx1250) — matrix operand order repaired and checked by reading, not compiled, not run on hardware
//
#include <hip/hip_runtime.h>
#include <hip/hip_bf16.h>

// ---------------------------------------------------------------------------
// MI455X / gfx1250 multi-head attention, bf16 WMMA path.
//   x[4,2048,1024] f32 -> out[4,2048,1024] f32
// Pipeline: cvt(bf16) -> QKV GEMM (WMMA) -> flash attention (WMMA) -> proj GEMM
// GEMM: 32M x 64N per wave (8 wmma / 12 b128 per k-step)
// Attention: 32 queries per wave, 32 keys per step (16 wmma / 16 b128)
// ---------------------------------------------------------------------------

typedef __bf16 bf16_t;
typedef __attribute__((ext_vector_type(16))) __bf16 v16bf;
typedef __attribute__((ext_vector_type(8)))  __bf16 v8bf;
typedef __attribute__((ext_vector_type(8)))  float  v8f;

#define B_SZ     4
#define T_LEN    2048
#define C_DIM    1024
#define NH       16
#define DK       64
#define C3       3072

// workspace byte offsets (total 75,497,472 B = 72 MiB)
#define XB_OFF   0u           // x bf16      [8192,1024]  16 MiB (reused as Y after attn)
#define WAB_OFF  16777216u    // W_attn bf16 [3072,1024]   6 MiB
#define WOB_OFF  23068672u    // W_o bf16    [1024,1024]   2 MiB
#define Q_OFF    25165824u    // Q bf16 [B,nh,T,dk]       16 MiB
#define K_OFF    41943040u    // K bf16 [B,nh,T,dk]       16 MiB
#define VT_OFF   58720256u    // V^T bf16 [B,nh,dk,T]     16 MiB
#define YB_OFF   XB_OFF       // attn output bf16 [8192,1024]

union V16U { v16bf v; v8bf h[2]; unsigned int u[8]; };

__device__ __forceinline__ v8f wmma_bf16(v16bf a, v16bf b, v8f c) {
  return __builtin_amdgcn_wmma_f32_16x16x32_bf16(false, a, false, b, (short)0, c,
                                                 false, false);
}

// A-layout 16x32 bf16 fragment: lane row fixed; element e = k = kk+2*hi8+e.
// hi8 = 8*(lane>=16).
__device__ __forceinline__ v16bf load_fragA(const bf16_t* rowp, int kk, int hi8) {
  V16U f;
  f.h[0] = *(const v8bf*)(rowp + kk + 2 * hi8);
  f.h[1] = *(const v8bf*)(rowp + kk + 2 * hi8 + 8);
  return f.v;
}

// B-layout 32x16 bf16 fragment: lane col fixed; element e = k = kk+hi16+e.
// hi16 = 16*(lane>=16).  32 contiguous bytes per lane.
__device__ __forceinline__ v16bf load_fragB(const bf16_t* rowp, int kk, int hi16) {
  V16U f;
  const v8bf* p = (const v8bf*)(rowp + kk + hi16);
  f.h[0] = p[0];
  f.h[1] = p[1];
  return f.v;
}

__device__ __forceinline__ unsigned int pk2(float a, float b) {
  union { bf16_t h[2]; unsigned int u; } t;
  t.h[0] = (bf16_t)a;
  t.h[1] = (bf16_t)b;
  return t.u;
}

// ---------------------------------------------------------------------------
// fp32 -> bf16 conversion, 4 elements/thread
// ---------------------------------------------------------------------------
__global__ void cvt_bf16_kernel(const float* __restrict__ src,
                                bf16_t* __restrict__ dst, int n) {
  int i = (blockIdx.x * blockDim.x + threadIdx.x) * 4;
  if (i < n) {
    float4 v = *(const float4*)(src + i);
    uint2 w;
    w.x = pk2(v.x, v.y);
    w.y = pk2(v.z, v.w);
    *(uint2*)(dst + i) = w;
  }
}

// ---------------------------------------------------------------------------
// QKV projection: qkv[m,n] = sum_k X[m,k]*W[n,k] + b[n],  M=8192 K=1024 N=3072
// Wave tile: 32M x 64N.  Scatter: Q,K -> [B,nh,T,dk]; V -> transposed [B,nh,dk,T]
// ---------------------------------------------------------------------------
__global__ __launch_bounds__(256) void qkv_gemm_kernel(
    const bf16_t* __restrict__ X, const bf16_t* __restrict__ W,
    const float* __restrict__ bias, bf16_t* __restrict__ Q,
    bf16_t* __restrict__ K, bf16_t* __restrict__ Vt) {
  const int lane = threadIdx.x & 31;
  const int wave = threadIdx.x >> 5;
  const int task = blockIdx.x * 8 + wave;     // 12288 wave tiles
  const int nt = task % 48;
  const int mt = task / 48;
  const int m0 = mt * 32;
  const int n0 = nt * 64;
  const int hi8  = (lane >> 4) << 3;
  const int hi16 = (lane >> 4) << 4;
  const int ln   = lane & 15;

  const bf16_t* arow[2];
#pragma unroll
  for (int mi = 0; mi < 2; ++mi)
    arow[mi] = X + (size_t)(m0 + mi * 16 + ln) * C_DIM;
  const bf16_t* brow[4];
#pragma unroll
  for (int j = 0; j < 4; ++j)
    brow[j] = W + (size_t)(n0 + j * 16 + ln) * C_DIM;

  v8f acc[2][4];
#pragma unroll
  for (int mi = 0; mi < 2; ++mi)
#pragma unroll
    for (int j = 0; j < 4; ++j)
#pragma unroll
      for (int r = 0; r < 8; ++r) acc[mi][j][r] = 0.0f;

  for (int kk = 0; kk < C_DIM; kk += 32) {
    v16bf fa0 = load_fragA(arow[0], kk, hi8);
    v16bf fa1 = load_fragA(arow[1], kk, hi8);
#pragma unroll
    for (int j = 0; j < 4; ++j) {
      v16bf fb = load_fragB(brow[j], kk, hi16);
      acc[0][j] = wmma_bf16(fa0, fb, acc[0][j]);
      acc[1][j] = wmma_bf16(fa1, fb, acc[1][j]);
    }
  }

#pragma unroll
  for (int mi = 0; mi < 2; ++mi) {
    const int mrow = m0 + mi * 16;
    const int bb = mrow >> 11;       // batch
    const int t0 = mrow & (T_LEN - 1);
#pragma unroll
    for (int j = 0; j < 4; ++j) {
      const int gn  = n0 + j * 16 + ln;
      const float bv = bias[gn];
      const int sec = gn >> 10;          // 0:Q 1:K 2:V (wave-uniform)
      const int rem = gn & 1023;
      const int hh  = rem >> 6;
      const int dd  = rem & 63;
      if (sec == 2) {
        // V^T[((b*16+hh)*64+dd)*T + t], t = t0+hi8+r contiguous -> one b128
        bf16_t* vp = Vt + ((size_t)((bb * NH + hh) * DK + dd)) * T_LEN + t0 + hi8;
        uint4 w;
        w.x = pk2(acc[mi][j][0] + bv, acc[mi][j][1] + bv);
        w.y = pk2(acc[mi][j][2] + bv, acc[mi][j][3] + bv);
        w.z = pk2(acc[mi][j][4] + bv, acc[mi][j][5] + bv);
        w.w = pk2(acc[mi][j][6] + bv, acc[mi][j][7] + bv);
        *(uint4*)vp = w;
      } else {
        bf16_t* op = (sec == 0 ? Q : K) +
                     ((size_t)((bb * NH + hh) * T_LEN) + t0 + hi8) * DK + dd;
#pragma unroll
        for (int r = 0; r < 8; ++r)
          op[(size_t)r * DK] = (bf16_t)(acc[mi][j][r] + bv);
      }
    }
  }
}

// ---------------------------------------------------------------------------
// Flash attention.  One wave = one (b, h, 32-query tile).  32 keys per step:
//   S^T = K * Q^T   (per query tile: two 16x16 C tiles)
//   online softmax along M (keys) -> per-lane stats after shfl_xor(16)
//   P^T rebuilt into B-layout with packs + shfl_xor(16)
//   O^T += V^T * P^T (V fragments shared by both query tiles)
// ---------------------------------------------------------------------------
__global__ __launch_bounds__(128) void attn_kernel(
    const bf16_t* __restrict__ Q, const bf16_t* __restrict__ K,
    const bf16_t* __restrict__ Vt, bf16_t* __restrict__ Y) {
  const int lane = threadIdx.x & 31;
  const int wave = threadIdx.x >> 5;
  const int task = blockIdx.x * 4 + wave;     // 4096 tasks
  const int qt = task & 63;                   // T/32
  const int hh = (task >> 6) & (NH - 1);
  const int bb = task >> 10;
  const int q0 = qt * 32;
  const int hi8  = (lane >> 4) << 3;
  const int hi16 = (lane >> 4) << 4;
  const int ln   = lane & 15;

  const bf16_t* Qh = Q  + (size_t)(bb * NH + hh) * T_LEN * DK;
  const bf16_t* Kh = K  + (size_t)(bb * NH + hh) * T_LEN * DK;
  const bf16_t* Vh = Vt + (size_t)(bb * NH + hh) * DK * T_LEN;

  v16bf fragQt[2][2];
#pragma unroll
  for (int t = 0; t < 2; ++t) {
    const bf16_t* qrow = Qh + (size_t)(q0 + t * 16 + ln) * DK;
    fragQt[t][0] = load_fragB(qrow, 0, hi16);   // dk chunk [0,32)
    fragQt[t][1] = load_fragB(qrow, 32, hi16);  // dk chunk [32,64)
  }

  v8f acc[2][4];
#pragma unroll
  for (int t = 0; t < 2; ++t)
#pragma unroll
    for (int c = 0; c < 4; ++c)
#pragma unroll
      for (int r = 0; r < 8; ++r) acc[t][c][r] = 0.0f;

  float m_run[2] = {-__builtin_inff(), -__builtin_inff()};
  float l_run[2] = {0.0f, 0.0f};
  const int qg[2] = {q0 + ln, q0 + 16 + ln};
  const int nsteps = q0 / 32 + 1;             // causal: keys <= q0+31

  for (int s = 0; s < nsteps; ++s) {
    const int kb = s * 32;
    // ---- S^T = K * Q^T for both query tiles (K fragments shared) ----
    v8f S[2][2];
#pragma unroll
    for (int t = 0; t < 2; ++t)
#pragma unroll
      for (int u = 0; u < 2; ++u)
#pragma unroll
        for (int r = 0; r < 8; ++r) S[t][u][r] = 0.0f;
    {
      const bf16_t* krow0 = Kh + (size_t)(kb + ln) * DK;
      const bf16_t* krow1 = Kh + (size_t)(kb + 16 + ln) * DK;
      const v16bf fk00 = load_fragA(krow0, 0, hi8);
      const v16bf fk01 = load_fragA(krow0, 32, hi8);
      const v16bf fk10 = load_fragA(krow1, 0, hi8);
      const v16bf fk11 = load_fragA(krow1, 32, hi8);
#pragma unroll
      for (int t = 0; t < 2; ++t) {
        S[t][0] = wmma_bf16(fk00, fragQt[t][0], S[t][0]);
        S[t][0] = wmma_bf16(fk01, fragQt[t][1], S[t][0]);
        S[t][1] = wmma_bf16(fk10, fragQt[t][0], S[t][1]);
        S[t][1] = wmma_bf16(fk11, fragQt[t][1], S[t][1]);
      }
    }
    // ---- online softmax + P^T relayout, per query tile ----
    V16U fp[2];
    const bool hi = (lane >= 16);
#pragma unroll
    for (int t = 0; t < 2; ++t) {
      const bool need_mask = (kb + 31 > q0 + t * 16);   // wave-uniform
      float mx = -__builtin_inff();
#pragma unroll
      for (int r = 0; r < 8; ++r) {
        float s0 = S[t][0][r] * 0.125f;                 // 1/sqrt(64)
        float s1 = S[t][1][r] * 0.125f;
        if (need_mask) {
          if (kb + hi8 + r > qg[t])      s0 = -__builtin_inff();
          if (kb + 16 + hi8 + r > qg[t]) s1 = -__builtin_inff();
        }
        S[t][0][r] = s0;
        S[t][1][r] = s1;
        mx = fmaxf(mx, fmaxf(s0, s1));
      }
      mx = fmaxf(mx, __shfl_xor(mx, 16));
      const float m_new = fmaxf(m_run[t], mx);
      const float sc = __expf(m_run[t] - m_new);
      float psum = 0.0f;
      unsigned int p0[4], p1[4];
#pragma unroll
      for (int j = 0; j < 4; ++j) {
        float a0 = __expf(S[t][0][2 * j] - m_new);
        float a1 = __expf(S[t][0][2 * j + 1] - m_new);
        float c0 = __expf(S[t][1][2 * j] - m_new);
        float c1 = __expf(S[t][1][2 * j + 1] - m_new);
        psum += (a0 + a1) + (c0 + c1);
        p0[j] = pk2(a0, a1);
        p1[j] = pk2(c0, c1);
      }
      psum += __shfl_xor(psum, 16);
      l_run[t] = l_run[t] * sc + psum;
      m_run[t] = m_new;
#pragma unroll
      for (int c = 0; c < 4; ++c)
#pragma unroll
        for (int r = 0; r < 8; ++r) acc[t][c][r] *= sc;
      // P^T (32 keys x 16 queries) C-layout -> B-layout: lanes<16 want keys
      // 0..15 (own M0..7 + partner M8..15), lanes>=16 want keys 16..31.
#pragma unroll
      for (int j = 0; j < 4; ++j) {
        unsigned int o0 = __shfl_xor(p0[j], 16);
        unsigned int o1 = __shfl_xor(p1[j], 16);
        fp[t].u[j]     = hi ? o1 : p0[j];
        fp[t].u[j + 4] = hi ? p1[j] : o0;
      }
    }
    // ---- O^T += V^T * P^T (V fragments shared by both query tiles) ----
    const bf16_t* vrow = Vh + (size_t)ln * T_LEN;   // row d = c*16 + ln
#pragma unroll
    for (int c = 0; c < 4; ++c) {
      const v16bf fv = load_fragA(vrow + (size_t)c * 16 * T_LEN, kb, hi8);
      acc[0][c] = wmma_bf16(fv, fp[0].v, acc[0][c]);
      acc[1][c] = wmma_bf16(fv, fp[1].v, acc[1][c]);
    }
  }

  // O^T C-layout: lane -> query q0+t*16+ln; d = c*16 + hi8 + r (r contiguous)
#pragma unroll
  for (int t = 0; t < 2; ++t) {
    const float inv = 1.0f / l_run[t];
    bf16_t* yrow = Y + ((size_t)(bb * T_LEN) + q0 + t * 16 + ln) * C_DIM + hh * DK;
#pragma unroll
    for (int c = 0; c < 4; ++c) {
      uint4 w;
      w.x = pk2(acc[t][c][0] * inv, acc[t][c][1] * inv);
      w.y = pk2(acc[t][c][2] * inv, acc[t][c][3] * inv);
      w.z = pk2(acc[t][c][4] * inv, acc[t][c][5] * inv);
      w.w = pk2(acc[t][c][6] * inv, acc[t][c][7] * inv);
      *(uint4*)(yrow + c * 16 + hi8) = w;
    }
  }
}

// ---------------------------------------------------------------------------
// Output projection: out[m,n] = sum_k Y[m,k]*Wo[n,k] + b_o[n] -> fp32 d_out
// Wave tile: 32M x 64N.
// ---------------------------------------------------------------------------
__global__ __launch_bounds__(256) void proj_gemm_kernel(
    const bf16_t* __restrict__ Y, const bf16_t* __restrict__ W,
    const float* __restrict__ bias, float* __restrict__ out) {
  const int lane = threadIdx.x & 31;
  const int wave = threadIdx.x >> 5;
  const int task = blockIdx.x * 8 + wave;     // 4096 wave tiles
  const int nt = task & 15;
  const int mt = task >> 4;
  const int m0 = mt * 32;
  const int n0 = nt * 64;
  const int hi8  = (lane >> 4) << 3;
  const int hi16 = (lane >> 4) << 4;
  const int ln   = lane & 15;

  const bf16_t* arow[2];
#pragma unroll
  for (int mi = 0; mi < 2; ++mi)
    arow[mi] = Y + (size_t)(m0 + mi * 16 + ln) * C_DIM;
  const bf16_t* brow[4];
#pragma unroll
  for (int j = 0; j < 4; ++j)
    brow[j] = W + (size_t)(n0 + j * 16 + ln) * C_DIM;

  v8f acc[2][4];
#pragma unroll
  for (int mi = 0; mi < 2; ++mi)
#pragma unroll
    for (int j = 0; j < 4; ++j)
#pragma unroll
      for (int r = 0; r < 8; ++r) acc[mi][j][r] = 0.0f;

  for (int kk = 0; kk < C_DIM; kk += 32) {
    v16bf fa0 = load_fragA(arow[0], kk, hi8);
    v16bf fa1 = load_fragA(arow[1], kk, hi8);
#pragma unroll
    for (int j = 0; j < 4; ++j) {
      v16bf fb = load_fragB(brow[j], kk, hi16);
      acc[0][j] = wmma_bf16(fa0, fb, acc[0][j]);
      acc[1][j] = wmma_bf16(fa1, fb, acc[1][j]);
    }
  }

#pragma unroll
  for (int mi = 0; mi < 2; ++mi) {
#pragma unroll
    for (int j = 0; j < 4; ++j) {
      const int gn = n0 + j * 16 + ln;
      const float bv = bias[gn];
      float* op = out + (size_t)(m0 + mi * 16 + hi8) * C_DIM + gn;
#pragma unroll
      for (int r = 0; r < 8; ++r) op[(size_t)r * C_DIM] = acc[mi][j][r] + bv;
    }
  }
}

// ---------------------------------------------------------------------------
extern "C" void kernel_launch(void* const* d_in, const int* in_sizes, int n_in,
                              void* d_out, int out_size, void* d_ws, size_t ws_size,
                              hipStream_t stream) {
  const float* x      = (const float*)d_in[0];
  const float* W_attn = (const float*)d_in[1];
  const float* b_attn = (const float*)d_in[2];
  const float* W_o    = (const float*)d_in[3];
  const float* b_o    = (const float*)d_in[4];
  float* out = (float*)d_out;

  char* ws = (char*)d_ws;
  bf16_t* Xb  = (bf16_t*)(ws + XB_OFF);
  bf16_t* Wab = (bf16_t*)(ws + WAB_OFF);
  bf16_t* Wob = (bf16_t*)(ws + WOB_OFF);
  bf16_t* Qb  = (bf16_t*)(ws + Q_OFF);
  bf16_t* Kb  = (bf16_t*)(ws + K_OFF);
  bf16_t* Vtb = (bf16_t*)(ws + VT_OFF);
  bf16_t* Yb  = (bf16_t*)(ws + YB_OFF);

  const int nx  = B_SZ * T_LEN * C_DIM;   // 8388608
  const int nwa = C3 * C_DIM;             // 3145728
  const int nwo = C_DIM * C_DIM;          // 1048576

  cvt_bf16_kernel<<<(nx / 4 + 255) / 256, 256, 0, stream>>>(x, Xb, nx);
  cvt_bf16_kernel<<<(nwa / 4 + 255) / 256, 256, 0, stream>>>(W_attn, Wab, nwa);
  cvt_bf16_kernel<<<(nwo / 4 + 255) / 256, 256, 0, stream>>>(W_o, Wob, nwo);

  // 12288 wave tiles / 8 waves per block
  qkv_gemm_kernel<<<1536, 256, 0, stream>>>(Xb, Wab, b_attn, Qb, Kb, Vtb);
  // 4096 (b,h,32-query) tasks / 4 waves per block
  attn_kernel<<<1024, 128, 0, stream>>>(Qb, Kb, Vtb, Yb);
  // 4096 wave tiles / 8 waves per block
  proj_gemm_kernel<<<512, 256, 0, stream>>>(Yb, Wob, b_o, out);
}
